// Model_67516885893191
// MI455X (gfx1250) — matrix operand order repaired and checked by reading, not compiled, not run on hardware
//
#include <hip/hip_runtime.h>
#include <hip/hip_bf16.h>
#include <math.h>

typedef __attribute__((ext_vector_type(16))) __bf16 v16bf;
typedef __attribute__((ext_vector_type(8)))  float  v8f;

#define F_DIM   36
#define HID     64
#define T_STEPS 256
#define B_TOTAL 4096
#define BT      16          // batch rows per block (one M tile)
#define KD      64          // padded K for delta (gamma_h GEMM, 36 -> 64)
#define KI      96          // padded K for [value_h|mask] GEMM (72 -> 96)
#define KH      64          // K for h GEMM

// d_ws layout: float[0]=sum_bce, float[1]=sum_w; bf16 weights at byte 64
#define WS_WEIGHT_OFF 64
#define WH_OFF   0                       // [64][KD]
#define WIH_OFF  (64*KD)                 // [192][KI]
#define WHH_OFF  (64*KD + 192*KI)        // [192][KH]
#define W_TOTAL  (64*KD + 192*KI + 192*KH)

// Native bf16 convert: gfx1250 has hardware bf16, so this lowers to a single
// v_cvt instruction instead of the ~6-VALU manual RNE bit-twiddle.
__device__ __forceinline__ unsigned short f2bf(float f) {
    return __builtin_bit_cast(unsigned short, (__bf16)f);
}

struct Frag32B { uint4 lo, hi; };

// A fragment (16-bit A 16x32, ISA 7.12.2): lane m=l&15, half=l>>4,
// VGPR0..3 <- K = 32*ks + 8*half + [0..7], VGPR4..7 <- +16.
__device__ __forceinline__ v16bf fragA_lds(const unsigned short* base, int strideE,
                                           int kstep, int lane) {
    const int m = lane & 15, half = lane >> 4;
    const int k0 = 32 * kstep + 8 * half;
    const unsigned short* p = base + m * strideE + k0;
    Frag32B f;
    f.lo = *reinterpret_cast<const uint4*>(p);
    f.hi = *reinterpret_cast<const uint4*>(p + 16);
    return __builtin_bit_cast(v16bf, f);
}

// B fragment (16-bit B 32x16): lane n=l&15, half=l>>4,
// VGPR0..3 <- K = 32*ks + 8*half + [0..7], VGPR4..7 <- +16; weights stored column-per-row (K contiguous).
__device__ __forceinline__ v16bf fragB_glb(const unsigned short* base, int strideE,
                                           int colbase, int kstep, int lane) {
    const int n = lane & 15, half = lane >> 4;
    const int k0 = 32 * kstep + 8 * half;
    const unsigned short* p = base + (size_t)(colbase + n) * strideE + k0;
    Frag32B f;
    f.lo = *reinterpret_cast<const uint4*>(p);
    f.hi = *reinterpret_cast<const uint4*>(p + 16);
    return __builtin_bit_cast(v16bf, f);
}

__global__ void grud_init(float* ws) {
    if (threadIdx.x < 2) ws[threadIdx.x] = 0.f;
}

__global__ void grud_prep(const float* __restrict__ Wh, const float* __restrict__ W_ih,
                          const float* __restrict__ W_hh, unsigned short* __restrict__ wbuf) {
    int i = blockIdx.x * 256 + threadIdx.x;
    if (i < 64 * KD) {                       // Wh: [64][36] -> [64][KD]
        int n = i / KD, k = i % KD;
        wbuf[WH_OFF + i] = (k < F_DIM) ? f2bf(Wh[n * F_DIM + k]) : (unsigned short)0;
    }
    int j = i - 64 * KD;
    if (j >= 0 && j < 192 * KI) {            // W_ih: [192][72] -> [192][KI]
        int n = j / KI, k = j % KI;
        wbuf[WIH_OFF + j] = (k < 2 * F_DIM) ? f2bf(W_ih[n * 2 * F_DIM + k]) : (unsigned short)0;
    }
    int k2 = i - (64 * KD + 192 * KI);
    if (k2 >= 0 && k2 < 192 * KH) {          // W_hh: [192][64] as-is
        wbuf[WHH_OFF + k2] = f2bf(W_hh[k2]);
    }
}

__global__ __launch_bounds__(128) void grud_main(
    const float* __restrict__ values, const float* __restrict__ masks,
    const float* __restrict__ deltas, const float* __restrict__ lasts,
    const float* __restrict__ label,  const float* __restrict__ is_train,
    const float* __restrict__ Wx,     const float* __restrict__ bx,
    const float* __restrict__ bh,
    const float* __restrict__ b_ih,   const float* __restrict__ b_hh,
    const float* __restrict__ W_out,  const float* __restrict__ b_out,
    const unsigned short* __restrict__ wbuf,
    float* __restrict__ out_imp, float* __restrict__ out_pred,
    float* __restrict__ ws_acc)
{
    __shared__ __align__(16) unsigned short sA_delta[BT * KD];
    __shared__ __align__(16) unsigned short sA_inp[BT * KI];
    __shared__ __align__(16) unsigned short sA_h[BT * KH];
    __shared__ float sDiagWx[F_DIM], sBx[F_DIM];
    __shared__ float sHfin[BT * HID];

    const int tid  = threadIdx.x;
    const int lane = tid & 31;
    const int wave = tid >> 5;                 // N tile 0..3
    const int batch0 = blockIdx.x * BT;

    for (int i = tid; i < BT * KD; i += 128) sA_delta[i] = 0;   // zero K padding once
    for (int i = tid; i < BT * KI; i += 128) sA_inp[i]   = 0;
    if (tid < F_DIM) { sDiagWx[tid] = Wx[tid * F_DIM + tid]; sBx[tid] = bx[tid]; }

    // per-lane output column + biases
    const int ncol = wave * 16 + (lane & 15);
    const float bh_l = bh[ncol];
    const float bir = b_ih[ncol],       bhr = b_hh[ncol];
    const float biz = b_ih[64 + ncol],  bhz = b_hh[64 + ncol];
    const float bin = b_ih[128 + ncol], bhn = b_hh[128 + ncol];

    // resident bf16 weight B-fragments (reused all 256 steps)
    const unsigned short* whB  = wbuf + WH_OFF;
    const unsigned short* wihB = wbuf + WIH_OFF;
    const unsigned short* whhB = wbuf + WHH_OFF;
    v16bf fWh[2], fWih[3][3], fWhh[3][2];
    #pragma unroll
    for (int ks = 0; ks < 2; ++ks) fWh[ks] = fragB_glb(whB, KD, wave * 16, ks, lane);
    #pragma unroll
    for (int g = 0; g < 3; ++g) {
        #pragma unroll
        for (int ks = 0; ks < 3; ++ks)
            fWih[g][ks] = fragB_glb(wihB, KI, g * 64 + wave * 16, ks, lane);
        #pragma unroll
        for (int ks = 0; ks < 2; ++ks)
            fWhh[g][ks] = fragB_glb(whhB, KH, g * 64 + wave * 16, ks, lane);
    }

    v8f h = {};   // hidden state, C layout, resident in accumulator registers
    __syncthreads();

    for (int t = 0; t < T_STEPS; ++t) {
        // ---- stage per-step inputs: impute + pack bf16 A matrices ----
        for (int i = tid; i < BT * F_DIM; i += 128) {
            const int row = i / F_DIM, f = i - row * F_DIM;
            const long long idx = ((long long)(batch0 + row) * T_STEPS + t) * F_DIM + f;
            const float v = values[idx], m = masks[idx], d = deltas[idx], l = lasts[idx];
            const float gx = __expf(-fmaxf(d * sDiagWx[f] + sBx[f], 0.f));
            const float vh = m * v + (1.f - m) * gx * l;
            out_imp[idx] = vh;
            sA_inp[row * KI + f]          = f2bf(vh);
            sA_inp[row * KI + F_DIM + f]  = f2bf(m);
            sA_delta[row * KD + f]        = f2bf(d);
            if (t + 1 < T_STEPS) {        // cover next step's latency
                __builtin_prefetch(&values[idx + F_DIM], 0, 1);
                __builtin_prefetch(&masks [idx + F_DIM], 0, 1);
                __builtin_prefetch(&deltas[idx + F_DIM], 0, 1);
                __builtin_prefetch(&lasts [idx + F_DIM], 0, 1);
            }
        }
        __syncthreads();

        // ---- gamma_h = exp(-relu(delta @ Wh^T + bh)); h *= gamma_h ----
        {
            v8f acc = {};
            #pragma unroll
            for (int ks = 0; ks < 2; ++ks) {
                v16bf a = fragA_lds(sA_delta, KD, ks, lane);
                acc = __builtin_amdgcn_wmma_f32_16x16x32_bf16(
                        false, a, false, fWh[ks], (short)0, acc, false, false);
            }
            #pragma unroll
            for (int r = 0; r < 8; ++r)
                h[r] *= __expf(-fmaxf(acc[r] + bh_l, 0.f));
        }
        // re-stage decayed h as bf16 A matrix
        {
            const int half = lane >> 4, nn = lane & 15;
            #pragma unroll
            for (int r = 0; r < 8; ++r)
                sA_h[(r + 8 * half) * KH + wave * 16 + nn] = f2bf(h[r]);
        }
        __syncthreads();

        // ---- gi = inp @ W_ih^T, gh = h @ W_hh^T (per-gate N tiles) ----
        v16bf ai[3], ah[2];
        #pragma unroll
        for (int ks = 0; ks < 3; ++ks) ai[ks] = fragA_lds(sA_inp, KI, ks, lane);
        #pragma unroll
        for (int ks = 0; ks < 2; ++ks) ah[ks] = fragA_lds(sA_h, KH, ks, lane);
        v8f gi[3], gh[3];
        #pragma unroll
        for (int g = 0; g < 3; ++g) {
            v8f a = {};
            #pragma unroll
            for (int ks = 0; ks < 3; ++ks)
                a = __builtin_amdgcn_wmma_f32_16x16x32_bf16(
                        false, ai[ks], false, fWih[g][ks], (short)0, a, false, false);
            gi[g] = a;
            v8f b = {};
            #pragma unroll
            for (int ks = 0; ks < 2; ++ks)
                b = __builtin_amdgcn_wmma_f32_16x16x32_bf16(
                        false, ah[ks], false, fWhh[g][ks], (short)0, b, false, false);
            gh[g] = b;
        }
        // ---- GRU gates, entirely in C-layout registers ----
        #pragma unroll
        for (int r = 0; r < 8; ++r) {
            const float rg = 1.f / (1.f + __expf(-(gi[0][r] + bir + gh[0][r] + bhr)));
            const float zg = 1.f / (1.f + __expf(-(gi[1][r] + biz + gh[1][r] + bhz)));
            const float ng = tanhf(gi[2][r] + bin + rg * (gh[2][r] + bhn));
            h[r] = (1.f - zg) * ng + zg * h[r];
        }
        __syncthreads();   // protect sA_* before next step's staging
    }

    // ---- head: logits, pred, BCE partial sums ----
    {
        const int half = lane >> 4, nn = lane & 15;
        #pragma unroll
        for (int r = 0; r < 8; ++r)
            sHfin[(r + 8 * half) * HID + wave * 16 + nn] = h[r];
    }
    __syncthreads();
    if (tid < BT) {
        float acc = b_out[0];
        for (int n = 0; n < HID; ++n) acc += sHfin[tid * HID + n] * W_out[n];
        const int bidx = batch0 + tid;
        out_pred[bidx] = 1.f / (1.f + __expf(-acc));
        const float y = label[bidx], w = is_train[bidx];
        const float bce = fmaxf(acc, 0.f) - acc * y + log1pf(__expf(-fabsf(acc)));
        atomicAdd(&ws_acc[0], bce * w);
        atomicAdd(&ws_acc[1], w);
    }
}

__global__ void grud_fin(const float* __restrict__ ws_acc, float* __restrict__ out0) {
    if (threadIdx.x == 0 && blockIdx.x == 0)
        out0[0] = ws_acc[0] / (ws_acc[1] + 1e-5f);
}

extern "C" void kernel_launch(void* const* d_in, const int* in_sizes, int n_in,
                              void* d_out, int out_size, void* d_ws, size_t ws_size,
                              hipStream_t stream) {
    (void)in_sizes; (void)n_in; (void)out_size; (void)ws_size;
    const float* values   = (const float*)d_in[0];
    const float* masks    = (const float*)d_in[1];
    const float* deltas   = (const float*)d_in[2];
    const float* lasts    = (const float*)d_in[3];
    const float* label    = (const float*)d_in[4];
    const float* is_train = (const float*)d_in[5];
    const float* Wx       = (const float*)d_in[6];
    const float* bx       = (const float*)d_in[7];
    const float* Wh       = (const float*)d_in[8];
    const float* bh       = (const float*)d_in[9];
    const float* W_ih     = (const float*)d_in[10];
    const float* W_hh     = (const float*)d_in[11];
    const float* b_ih     = (const float*)d_in[12];
    const float* b_hh     = (const float*)d_in[13];
    const float* W_out    = (const float*)d_in[14];
    const float* b_out    = (const float*)d_in[15];

    float* out = (float*)d_out;
    float* out_imp  = out + 1;                                            // [B,T,F]
    float* out_pred = out + 1 + (size_t)B_TOTAL * T_STEPS * F_DIM;        // [B]

    float* ws_acc = (float*)d_ws;
    unsigned short* wbuf = (unsigned short*)((char*)d_ws + WS_WEIGHT_OFF);

    grud_init<<<1, 32, 0, stream>>>(ws_acc);
    grud_prep<<<(W_TOTAL + 255) / 256, 256, 0, stream>>>(Wh, W_ih, W_hh, wbuf);
    grud_main<<<B_TOTAL / BT, 128, 0, stream>>>(
        values, masks, deltas, lasts, label, is_train,
        Wx, bx, bh, b_ih, b_hh, W_out, b_out,
        wbuf, out_imp, out_pred, ws_acc);
    grud_fin<<<1, 32, 0, stream>>>(ws_acc, out);
}
